// HofmanSwarm_52080773431952
// MI455X (gfx1250) — compile-verified, matrix operand order checked by reading, not run on hardware
//
#include <hip/hip_runtime.h>
#include <hip/hip_bf16.h>

// ---------------- problem constants ----------------
#define BB 8
#define TT 128
#define VV 32000
#define AA 64
#define DD 256
#define NSNS 16
#define SS 128
#define HH 4
#define HD 64
#define D2 512
#define NSD 4096
#define SCALE 8.0f

typedef __attribute__((ext_vector_type(16))) _Float16 v16h;
typedef __attribute__((ext_vector_type(8)))  _Float16 v8h;
typedef __attribute__((ext_vector_type(8)))  float    v8f;

__device__ __forceinline__ float sigmoid_f(float x) { return 1.0f / (1.0f + __expf(-x)); }
__device__ __forceinline__ float gelu_f(float x) { return 0.5f * x * (1.0f + erff(x * 0.70710678118654752f)); }

// ---------------- WMMA multi-tile core -------------
// A: f16 row-major [M x K] (row stride lda), rows clamped to M-1 (padding rows
//    produce garbage that is never stored).
// BT: f16 "transposed" [N x K] layout (row stride ldb = K).
// One A fragment feeds NT consecutive 16-wide column tiles (independent
// accumulators -> back-to-back WMMAs, no D->A/B hazard).
// Fragment layout per ISA 7.12.2 (16-bit A 16x32): lane L<16 -> M=L, K-runs
// {koff..koff+7, 16+koff..16+koff+7} with koff = (lane>>4)*8. B symmetric.
template <int NT>
__device__ __forceinline__ void wmma_tiles(const _Float16* __restrict__ A, long long lda, int M, int row0,
                                           const _Float16* __restrict__ BT, long long ldb, int col0,
                                           int k_begin, int k_end, v8f* acc) {
  const int lane = threadIdx.x & 31;
  int m = row0 + (lane & 15);
  m = (m < M) ? m : (M - 1);
  const int koff = (lane >> 4) << 3;  // 0 or 8
  const _Float16* __restrict__ arow = A + (long long)m * lda;
  const _Float16* brow[NT];
  #pragma unroll
  for (int i = 0; i < NT; ++i)
    brow[i] = BT + (long long)(col0 + i * 16 + (lane & 15)) * ldb;
  for (int k0 = k_begin; k0 < k_end; k0 += 32) {
    const v8h a0 = *reinterpret_cast<const v8h*>(arow + k0 + koff);
    const v8h a1 = *reinterpret_cast<const v8h*>(arow + k0 + koff + 16);
    v16h a;
    #pragma unroll
    for (int j = 0; j < 8; ++j) { a[j] = a0[j]; a[8 + j] = a1[j]; }
    #pragma unroll
    for (int i = 0; i < NT; ++i) {
      __builtin_prefetch(brow[i] + k0 + 256, 0, 1);  // stream next weight tiles
      const v8h b0 = *reinterpret_cast<const v8h*>(brow[i] + k0 + koff);
      const v8h b1 = *reinterpret_cast<const v8h*>(brow[i] + k0 + koff + 16);
      v16h b;
      #pragma unroll
      for (int j = 0; j < 8; ++j) { b[j] = b0[j]; b[8 + j] = b1[j]; }
      acc[i] = __builtin_amdgcn_wmma_f32_16x16x32_f16(false, a, false, b, (short)0, acc[i], false, false);
    }
  }
}

// C/D layout: VGPR r holds row = row0 + (lane>>4)*8 + r, col = col0 + (lane&15).
// Row-validity is half-wave-uniform for all shapes here (M multiple of 16, or
// M==8 where the upper-half rows are entirely padding), so guard once on rbase
// and emit the 8 stores unconditionally -> single store clause, no exec churn.
__device__ __forceinline__ void store_tile_f32(v8f acc, float* __restrict__ dst, long long ldd, int M,
                                               int row0, int col0, const float* __restrict__ bias, int act) {
  const int lane = threadIdx.x & 31;
  const int n = col0 + (lane & 15);
  const int rbase = row0 + ((lane >> 4) << 3);
  const float bv = bias ? bias[n] : 0.0f;
  if (rbase < M) {
    #pragma unroll
    for (int r = 0; r < 8; ++r) {
      float v = acc[r] + bv;
      if (act == 1) v = gelu_f(v);
      else if (act == 2) v = sigmoid_f(v);
      dst[(long long)(rbase + r) * ldd + n] = v;
    }
  }
}

__device__ __forceinline__ void store_tile_f16(v8f acc, _Float16* __restrict__ dst, long long ldd, int M,
                                               int row0, int col0) {
  const int lane = threadIdx.x & 31;
  const int n = col0 + (lane & 15);
  const int rbase = row0 + ((lane >> 4) << 3);
  if (rbase < M) {
    #pragma unroll
    for (int r = 0; r < 8; ++r)
      dst[(long long)(rbase + r) * ldd + n] = (_Float16)acc[r];
  }
}

// ---------------- setup kernels ----------------
__global__ void k_transpose_h(const float* __restrict__ src, _Float16* __restrict__ dst, int K, int N) {
  long long i = (long long)blockIdx.x * blockDim.x + threadIdx.x;
  if (i >= (long long)K * N) return;
  int k = (int)(i / N), n = (int)(i % N);
  dst[(long long)n * K + k] = (_Float16)src[i];
}

__global__ void k_cast_h(const float* __restrict__ src, _Float16* __restrict__ dst, int n) {
  int i = blockIdx.x * blockDim.x + threadIdx.x;
  if (i < n) dst[i] = (_Float16)src[i];
}

__global__ void k_zero(float* __restrict__ p, int n) {
  int i = blockIdx.x * blockDim.x + threadIdx.x;
  if (i < n) p[i] = 0.0f;
}

__global__ void k_zero_h(_Float16* __restrict__ p, int n) {
  int i = blockIdx.x * blockDim.x + threadIdx.x;
  if (i < n) p[i] = (_Float16)0.0f;
}

__global__ void k_embed(const int* __restrict__ idx, const float* __restrict__ emb,
                        const float* __restrict__ pos, float* __restrict__ percepts) {
  int i = blockIdx.x * blockDim.x + threadIdx.x;  // B*T*D
  if (i >= BB * TT * DD) return;
  int b = i / (TT * DD), r = i % (TT * DD), t = r / DD, d = r % DD;
  percepts[i] = emb[(long long)idx[b * TT + t] * DD + d] + pos[t * DD + d];
}

__global__ void k_pre(const float* __restrict__ bind, const float* __restrict__ aimp,
                      const float* __restrict__ cmix, float* __restrict__ sigbind,
                      float* __restrict__ iw, float* __restrict__ cw) {
  int i = blockIdx.x * blockDim.x + threadIdx.x;
  if (i < NSD) sigbind[i] = sigmoid_f(bind[i]);
  if (i < AA) {
    float mx = -1e30f;
    for (int j = 0; j < AA; ++j) mx = fmaxf(mx, aimp[j]);
    float s = 0.0f;
    for (int j = 0; j < AA; ++j) s += __expf(aimp[j] - mx);
    iw[i] = __expf(aimp[i] - mx) / s;
  }
  if (i < 2) {
    float mx = fmaxf(cmix[0], cmix[1]);
    float s = __expf(cmix[0] - mx) + __expf(cmix[1] - mx);
    cw[i] = __expf(cmix[i] - mx) / s;
  }
}

// ---------------- per-step kernels ----------------
__global__ void k_fbinit(float* __restrict__ fb, const float* __restrict__ bfb) {
  int i = blockIdx.x * blockDim.x + threadIdx.x;  // B*D
  if (i < BB * DD) fb[i] = bfb[i % DD];
}

// fb += prev_logits @ Wfb   (M=8,N=256,K=32000; K split over blockIdx.y, 4 tiles/wave)
__global__ void k_fb_gemm(const _Float16* __restrict__ logh, const _Float16* __restrict__ WfbT,
                          float* __restrict__ fb) {
  const int KC = VV / 25;  // 1280, multiple of 32
  v8f acc[4] = {};
  const int col0 = blockIdx.x * 64;
  wmma_tiles<4>(logh, VV, BB, 0, WfbT, VV, col0, blockIdx.y * KC, (blockIdx.y + 1) * KC, acc);
  const int lane = threadIdx.x & 31;
  const int rbase = (lane >> 4) << 3;
  if (rbase < BB) {
    #pragma unroll
    for (int i = 0; i < 4; ++i) {
      const int n = col0 + i * 16 + (lane & 15);
      #pragma unroll
      for (int r = 0; r < 8; ++r)
        atomicAdd(&fb[(rbase + r) * DD + n], acc[i][r]);
    }
  }
}

__global__ void k_fgmod(const float* __restrict__ fb, const float* __restrict__ percepts,
                        const float* __restrict__ Wfg, const float* __restrict__ bfg,
                        _Float16* __restrict__ mod_h, int t) {
  int i = blockIdx.x * blockDim.x + threadIdx.x;  // B*D
  if (i >= BB * DD) return;
  int b = i / DD, d = i % DD;
  const float* fbr = fb + b * DD;
  const float* pt = percepts + ((long long)b * TT + t) * DD;
  float acc = bfg[d];
  for (int j = 0; j < DD; ++j) acc += fbr[j] * Wfg[j * DD + d];
  for (int j = 0; j < DD; ++j) acc += pt[j] * Wfg[(DD + j) * DD + d];
  mod_h[i] = (_Float16)(pt[d] * (1.0f + sigmoid_f(acc)));
}

// agents[:, :NS] += reshape(mod @ Wsens + bsens) * sigbind   (M=8,N=4096,K=256)
__global__ void k_sens(const _Float16* __restrict__ mod_h, const _Float16* __restrict__ WsensT,
                       const float* __restrict__ bsens, const float* __restrict__ sigbind,
                       float* __restrict__ agents) {
  v8f acc[4] = {};
  const int col0 = blockIdx.x * 64;
  wmma_tiles<4>(mod_h, DD, BB, 0, WsensT, DD, col0, 0, DD, acc);
  const int lane = threadIdx.x & 31;
  const int rbase = (lane >> 4) << 3;
  if (rbase < BB) {
    #pragma unroll
    for (int i = 0; i < 4; ++i) {
      const int n = col0 + i * 16 + (lane & 15);  // [0,4096)
      const int slot = n >> 8, d = n & 255;
      const float bsv = bsens[n], sbv = sigbind[n];
      #pragma unroll
      for (int r = 0; r < 8; ++r)
        agents[((long long)(rbase + r) * AA + slot) * DD + d] += (acc[i][r] + bsv) * sbv;
    }
  }
}

__global__ void k_agentsT(const float* __restrict__ agents, _Float16* __restrict__ agentsT) {
  int i = blockIdx.x * blockDim.x + threadIdx.x;  // B*D*A
  if (i >= BB * DD * AA) return;
  int b = i / (DD * AA), r = i % (DD * AA), n = r / AA, k = r % AA;
  agentsT[i] = (_Float16)agents[((long long)b * AA + k) * DD + n];
}

// inter[b] = anet[c] @ agents[b]  (M=64,N=256,K=64 per batch) -> f16
__global__ void k_inter(const _Float16* __restrict__ anet_c, const _Float16* __restrict__ agentsT,
                        _Float16* __restrict__ inter_h) {
  const int b = blockIdx.z;
  v8f acc[4] = {};
  const int col0 = blockIdx.x * 64;
  wmma_tiles<4>(anet_c, AA, AA, blockIdx.y * 16, agentsT + (long long)b * DD * AA, AA, col0, 0, AA, acc);
  #pragma unroll
  for (int i = 0; i < 4; ++i)
    store_tile_f16(acc[i], inter_h + (long long)b * AA * DD, DD, AA, blockIdx.y * 16, col0 + i * 16);
}

__global__ void k_cat(const float* __restrict__ agents, const _Float16* __restrict__ inter_h,
                      _Float16* __restrict__ cat_h) {
  int i = blockIdx.x * blockDim.x + threadIdx.x;  // B*A*2D
  if (i >= BB * AA * D2) return;
  int r = i / D2, j = i % D2;
  cat_h[i] = (j < DD) ? (_Float16)agents[r * DD + j] : inter_h[r * DD + (j - DD)];
}

// cand = gelu(inter @ Wc[c] + bc[c])  (M=512,N=256,K=256)
__global__ void k_cand(const _Float16* __restrict__ inter_h, const _Float16* __restrict__ WcT_c,
                       const float* __restrict__ bc_c, float* __restrict__ cand) {
  v8f acc[4] = {};
  const int col0 = blockIdx.x * 64;
  wmma_tiles<4>(inter_h, DD, BB * AA, blockIdx.y * 16, WcT_c, DD, col0, 0, DD, acc);
  #pragma unroll
  for (int i = 0; i < 4; ++i)
    store_tile_f32(acc[i], cand, DD, BB * AA, blockIdx.y * 16, col0 + i * 16, bc_c, 1);
}

// gate = sigmoid(cat @ Wg[c] + bg[c])  (M=512,N=256,K=512)
__global__ void k_gate(const _Float16* __restrict__ cat_h, const _Float16* __restrict__ WgT_c,
                       const float* __restrict__ bg_c, float* __restrict__ gate) {
  v8f acc[4] = {};
  const int col0 = blockIdx.x * 64;
  wmma_tiles<4>(cat_h, D2, BB * AA, blockIdx.y * 16, WgT_c, D2, col0, 0, D2, acc);
  #pragma unroll
  for (int i = 0; i < 4; ++i)
    store_tile_f32(acc[i], gate, DD, BB * AA, blockIdx.y * 16, col0 + i * 16, bg_c, 2);
}

// per-row LayerNorm mix; c==0 writes newagents, c==1 writes agents
__global__ void k_mixln(const float* __restrict__ gate, const float* __restrict__ cand,
                        float* __restrict__ agents, float* __restrict__ newagents,
                        const float* __restrict__ lng, const float* __restrict__ lnb,
                        const float* __restrict__ cw, int c) {
  const int r = blockIdx.x, d = threadIdx.x;
  __shared__ float red[DD];
  const float g = gate[r * DD + d], cd = cand[r * DD + d], ag = agents[r * DD + d];
  const float x = g * cd + (1.0f - g) * ag;
  red[d] = x; __syncthreads();
  for (int o = 128; o > 0; o >>= 1) { if (d < o) red[d] += red[d + o]; __syncthreads(); }
  const float mean = red[0] / DD; __syncthreads();
  const float xc = x - mean;
  red[d] = xc * xc; __syncthreads();
  for (int o = 128; o > 0; o >>= 1) { if (d < o) red[d] += red[d + o]; __syncthreads(); }
  const float var = red[0] / DD;
  const float y = xc * rsqrtf(var + 1e-5f) * lng[c * DD + d] + lnb[c * DD + d];
  if (c == 0) newagents[r * DD + d] = cw[0] * y;
  else        agents[r * DD + d] = newagents[r * DD + d] + cw[1] * y;
}

__global__ void k_cons(const float* __restrict__ agents, const float* __restrict__ iw,
                       float* __restrict__ cons) {
  int i = blockIdx.x * blockDim.x + threadIdx.x;  // B*D
  if (i >= BB * DD) return;
  int b = i / DD, d = i % DD;
  float acc = 0.0f;
  for (int a = 0; a < AA; ++a) acc += agents[((long long)b * AA + a) * DD + d] * iw[a];
  cons[i] = acc;
}

__global__ void k_q(const float* __restrict__ cons, const float* __restrict__ Wq,
                    const float* __restrict__ bq, float* __restrict__ qb) {
  int i = blockIdx.x * blockDim.x + threadIdx.x;  // B*D
  if (i >= BB * DD) return;
  int b = i / DD, d = i % DD;
  float acc = bq[d];
  for (int j = 0; j < DD; ++j) acc += cons[b * DD + j] * Wq[j * DD + d];
  qb[i] = acc;
}

// K/V = mem @ W + b  (M=1024,N=256,K=256)
__global__ void k_kv(const _Float16* __restrict__ mem_h, const _Float16* __restrict__ WT,
                     const float* __restrict__ bias, float* __restrict__ dst) {
  v8f acc[4] = {};
  const int col0 = blockIdx.x * 64;
  wmma_tiles<4>(mem_h, DD, BB * SS, blockIdx.y * 16, WT, DD, col0, 0, DD, acc);
  #pragma unroll
  for (int i = 0; i < 4; ++i)
    store_tile_f32(acc[i], dst, DD, BB * SS, blockIdx.y * 16, col0 + i * 16, bias, 0);
}

__global__ void k_attn(const float* __restrict__ qb, const float* __restrict__ kbuf,
                       const float* __restrict__ vbuf, float* __restrict__ rec, int t) {
  const int b = blockIdx.x, h = blockIdx.y, s = threadIdx.x;  // 128 threads
  __shared__ float sc[SS], red[SS];
  const float* qv = qb + b * DD + h * HD;
  const float* kv = kbuf + ((long long)b * SS + s) * DD + h * HD;
  float score = 0.0f;
  for (int j = 0; j < HD; ++j) score += qv[j] * kv[j];
  score /= SCALE;
  if (s >= t) score = -1e9f;
  red[s] = score; __syncthreads();
  for (int o = 64; o > 0; o >>= 1) { if (s < o) red[s] = fmaxf(red[s], red[s + o]); __syncthreads(); }
  const float mx = red[0]; __syncthreads();
  const float e = __expf(score - mx);
  sc[s] = e; red[s] = e; __syncthreads();
  for (int o = 64; o > 0; o >>= 1) { if (s < o) red[s] += red[s + o]; __syncthreads(); }
  const float inv = 1.0f / red[0];
  __syncthreads();
  if (s < HD) {
    float acc = 0.0f;
    for (int j = 0; j < SS; ++j) acc += sc[j] * vbuf[((long long)b * SS + j) * DD + h * HD + s];
    rec[b * DD + h * HD + s] = acc * inv;
  }
}

__global__ void k_mgate(const float* __restrict__ rec, const float* __restrict__ Wo,
                        const float* __restrict__ bo, const float* __restrict__ Wmg,
                        const float* __restrict__ bmg, const float* __restrict__ mlng,
                        const float* __restrict__ mlnb, float* __restrict__ cons) {
  const int b = blockIdx.x, d = threadIdx.x;  // 256 threads
  __shared__ float r2[DD], cs[DD], red[DD];
  float acc = bo[d];
  const float* rb = rec + b * DD;
  for (int j = 0; j < DD; ++j) acc += rb[j] * Wo[j * DD + d];
  r2[d] = acc;
  cs[d] = cons[b * DD + d];
  __syncthreads();
  float g = bmg[d];
  for (int j = 0; j < DD; ++j) g += cs[j] * Wmg[j * DD + d];
  for (int j = 0; j < DD; ++j) g += r2[j] * Wmg[(DD + j) * DD + d];
  g = sigmoid_f(g);
  const float x = cs[d] + g * r2[d];
  red[d] = x; __syncthreads();
  for (int o = 128; o > 0; o >>= 1) { if (d < o) red[d] += red[d + o]; __syncthreads(); }
  const float mean = red[0] / DD; __syncthreads();
  const float xc = x - mean;
  red[d] = xc * xc; __syncthreads();
  for (int o = 128; o > 0; o >>= 1) { if (d < o) red[d] += red[d + o]; __syncthreads(); }
  const float var = red[0] / DD;
  cons[b * DD + d] = xc * rsqrtf(var + 1e-5f) * mlng[d] + mlnb[d];
}

__global__ void k_memwrite(const float* __restrict__ cons, _Float16* __restrict__ mem_h, int t) {
  int i = blockIdx.x * blockDim.x + threadIdx.x;  // B*D
  if (i >= BB * DD) return;
  int b = i / DD, d = i % DD;
  mem_h[((long long)b * SS + t) * DD + d] = (_Float16)cons[i];
}

__global__ void k_h(const float* __restrict__ cons, const float* __restrict__ Wa1,
                    const float* __restrict__ ba1, _Float16* __restrict__ hb_h) {
  int i = blockIdx.x * blockDim.x + threadIdx.x;  // B*2D
  if (i >= BB * D2) return;
  int b = i / D2, j = i % D2;
  float acc = ba1[j];
  for (int k = 0; k < DD; ++k) acc += cons[b * DD + k] * Wa1[k * D2 + j];
  hb_h[i] = (_Float16)gelu_f(acc);
}

// logits -> d_out slice (f32, row stride T*V) AND f16 copy for next step's fb GEMM
__global__ void k_logits(const _Float16* __restrict__ hb_h, const _Float16* __restrict__ Wa2T,
                         const float* __restrict__ ba2, float* __restrict__ out_t,
                         _Float16* __restrict__ logh) {
  v8f acc[4] = {};
  const int col0 = blockIdx.x * 64;
  wmma_tiles<4>(hb_h, D2, BB, 0, Wa2T, D2, col0, 0, D2, acc);
  const int lane = threadIdx.x & 31;
  const int rbase = (lane >> 4) << 3;
  if (rbase < BB) {
    #pragma unroll
    for (int i = 0; i < 4; ++i) {
      const int n = col0 + i * 16 + (lane & 15);
      const float bv = ba2[n];
      #pragma unroll
      for (int r = 0; r < 8; ++r) {
        const float v = acc[i][r] + bv;
        out_t[(long long)(rbase + r) * TT * VV + n] = v;
        logh[(long long)(rbase + r) * VV + n] = (_Float16)v;
      }
    }
  }
}

// ---------------- host ----------------
extern "C" void kernel_launch(void* const* d_in, const int* in_sizes, int n_in,
                              void* d_out, int out_size, void* d_ws, size_t ws_size,
                              hipStream_t stream) {
  const int*   idx   = (const int*)d_in[0];
  const float* emb   = (const float*)d_in[1];
  const float* pos   = (const float*)d_in[2];
  const float* Wsens = (const float*)d_in[3];
  const float* bsens = (const float*)d_in[4];
  const float* Wfb   = (const float*)d_in[5];
  const float* bfb   = (const float*)d_in[6];
  const float* Wfg   = (const float*)d_in[7];
  const float* bfg   = (const float*)d_in[8];
  const float* bind  = (const float*)d_in[9];
  const float* anet  = (const float*)d_in[10];
  const float* Wg    = (const float*)d_in[11];
  const float* bg    = (const float*)d_in[12];
  const float* Wc    = (const float*)d_in[13];
  const float* bc    = (const float*)d_in[14];
  const float* lng   = (const float*)d_in[15];
  const float* lnb   = (const float*)d_in[16];
  const float* cmix  = (const float*)d_in[17];
  const float* aimp  = (const float*)d_in[18];
  const float* Wq    = (const float*)d_in[19];
  const float* bq    = (const float*)d_in[20];
  const float* Wk    = (const float*)d_in[21];
  const float* bk    = (const float*)d_in[22];
  const float* Wv    = (const float*)d_in[23];
  const float* bv    = (const float*)d_in[24];
  const float* Wo    = (const float*)d_in[25];
  const float* bo    = (const float*)d_in[26];
  const float* Wmg   = (const float*)d_in[27];
  const float* bmg   = (const float*)d_in[28];
  const float* mlng  = (const float*)d_in[29];
  const float* mlnb  = (const float*)d_in[30];
  const float* Wa1   = (const float*)d_in[31];
  const float* ba1   = (const float*)d_in[32];
  const float* Wa2   = (const float*)d_in[33];
  const float* ba2   = (const float*)d_in[34];
  float* out = (float*)d_out;

  // bump allocator over d_ws
  char* base = (char*)d_ws;
  size_t off = 0;
  auto alloc = [&](size_t bytes) -> char* {
    char* p = base + off;
    off = (off + bytes + 255) & ~(size_t)255;
    return p;
  };
  float* percepts  = (float*)alloc((size_t)BB * TT * DD * 4);
  float* agents    = (float*)alloc((size_t)BB * AA * DD * 4);
  float* newagents = (float*)alloc((size_t)BB * AA * DD * 4);
  float* fb        = (float*)alloc((size_t)BB * DD * 4);
  float* cand      = (float*)alloc((size_t)BB * AA * DD * 4);
  float* gate      = (float*)alloc((size_t)BB * AA * DD * 4);
  float* cons      = (float*)alloc((size_t)BB * DD * 4);
  float* qb        = (float*)alloc((size_t)BB * DD * 4);
  float* kbuf      = (float*)alloc((size_t)BB * SS * DD * 4);
  float* vbuf      = (float*)alloc((size_t)BB * SS * DD * 4);
  float* rec       = (float*)alloc((size_t)BB * DD * 4);
  float* sigbind   = (float*)alloc((size_t)NSD * 4);
  float* iw        = (float*)alloc((size_t)AA * 4);
  float* cw        = (float*)alloc((size_t)2 * 4);
  _Float16* mod_h   = (_Float16*)alloc((size_t)BB * DD * 2);
  _Float16* inter_h = (_Float16*)alloc((size_t)BB * AA * DD * 2);
  _Float16* cat_h   = (_Float16*)alloc((size_t)BB * AA * D2 * 2);
  _Float16* mem_h   = (_Float16*)alloc((size_t)BB * SS * DD * 2);
  _Float16* hb_h    = (_Float16*)alloc((size_t)BB * D2 * 2);
  _Float16* logh    = (_Float16*)alloc((size_t)BB * VV * 2);
  _Float16* anet_h  = (_Float16*)alloc((size_t)2 * AA * AA * 2);
  _Float16* agentsT = (_Float16*)alloc((size_t)BB * DD * AA * 2);
  _Float16* WfbT   = (_Float16*)alloc((size_t)VV * DD * 2);
  _Float16* WsensT = (_Float16*)alloc((size_t)DD * NSD * 2);
  _Float16* WcT    = (_Float16*)alloc((size_t)2 * DD * DD * 2);
  _Float16* WgT    = (_Float16*)alloc((size_t)2 * D2 * DD * 2);
  _Float16* WkT    = (_Float16*)alloc((size_t)DD * DD * 2);
  _Float16* WvT    = (_Float16*)alloc((size_t)DD * DD * 2);
  _Float16* Wa2T   = (_Float16*)alloc((size_t)D2 * VV * 2);

  // ---- one-time setup (per call; deterministic) ----
  k_transpose_h<<<(VV * DD + 255) / 256, 256, 0, stream>>>(Wfb, WfbT, VV, DD);
  k_transpose_h<<<(DD * NSD + 255) / 256, 256, 0, stream>>>(Wsens, WsensT, DD, NSD);
  for (int c = 0; c < 2; ++c) {
    k_transpose_h<<<(DD * DD + 255) / 256, 256, 0, stream>>>(Wc + c * DD * DD, WcT + c * DD * DD, DD, DD);
    k_transpose_h<<<(D2 * DD + 255) / 256, 256, 0, stream>>>(Wg + c * D2 * DD, WgT + c * D2 * DD, D2, DD);
  }
  k_transpose_h<<<(DD * DD + 255) / 256, 256, 0, stream>>>(Wk, WkT, DD, DD);
  k_transpose_h<<<(DD * DD + 255) / 256, 256, 0, stream>>>(Wv, WvT, DD, DD);
  k_transpose_h<<<((long long)D2 * VV + 255) / 256, 256, 0, stream>>>(Wa2, Wa2T, D2, VV);
  k_cast_h<<<(2 * AA * AA + 255) / 256, 256, 0, stream>>>(anet, anet_h, 2 * AA * AA);

  k_zero<<<(BB * AA * DD + 255) / 256, 256, 0, stream>>>(agents, BB * AA * DD);
  k_zero_h<<<(BB * SS * DD + 255) / 256, 256, 0, stream>>>(mem_h, BB * SS * DD);
  k_pre<<<(NSD + 255) / 256, 256, 0, stream>>>(bind, aimp, cmix, sigbind, iw, cw);
  k_embed<<<(BB * TT * DD + 255) / 256, 256, 0, stream>>>(idx, emb, pos, percepts);

  // ---- sequential scan ----
  for (int t = 0; t < TT; ++t) {
    k_fbinit<<<(BB * DD + 255) / 256, 256, 0, stream>>>(fb, bfb);
    if (t > 0)
      k_fb_gemm<<<dim3(DD / 64, 25), 32, 0, stream>>>(logh, WfbT, fb);
    k_fgmod<<<(BB * DD + 255) / 256, 256, 0, stream>>>(fb, percepts, Wfg, bfg, mod_h, t);
    k_sens<<<NSD / 64, 32, 0, stream>>>(mod_h, WsensT, bsens, sigbind, agents);
    k_agentsT<<<(BB * DD * AA + 255) / 256, 256, 0, stream>>>(agents, agentsT);
    for (int c = 0; c < 2; ++c) {
      k_inter<<<dim3(DD / 64, AA / 16, BB), 32, 0, stream>>>(anet_h + c * AA * AA, agentsT, inter_h);
      k_cat<<<(BB * AA * D2 + 255) / 256, 256, 0, stream>>>(agents, inter_h, cat_h);
      k_cand<<<dim3(DD / 64, BB * AA / 16), 32, 0, stream>>>(inter_h, WcT + c * DD * DD, bc + c * DD, cand);
      k_gate<<<dim3(DD / 64, BB * AA / 16), 32, 0, stream>>>(cat_h, WgT + c * D2 * DD, bg + c * DD, gate);
      k_mixln<<<BB * AA, DD, 0, stream>>>(gate, cand, agents, newagents, lng, lnb, cw, c);
    }
    k_cons<<<(BB * DD + 255) / 256, 256, 0, stream>>>(agents, iw, cons);
    if (t > 0) {
      k_q<<<(BB * DD + 255) / 256, 256, 0, stream>>>(cons, Wq, bq, qb);
      k_kv<<<dim3(DD / 64, BB * SS / 16), 32, 0, stream>>>(mem_h, WkT, bk, kbuf);
      k_kv<<<dim3(DD / 64, BB * SS / 16), 32, 0, stream>>>(mem_h, WvT, bv, vbuf);
      k_attn<<<dim3(BB, HH), SS, 0, stream>>>(qb, kbuf, vbuf, rec, t);
      k_mgate<<<BB, DD, 0, stream>>>(rec, Wo, bo, Wmg, bmg, mlng, mlnb, cons);
    }
    k_memwrite<<<(BB * DD + 255) / 256, 256, 0, stream>>>(cons, mem_h, t);
    k_h<<<(BB * D2 + 255) / 256, 256, 0, stream>>>(cons, Wa1, ba1, hb_h);
    k_logits<<<VV / 64, 32, 0, stream>>>(hb_h, Wa2T, ba2, out + (size_t)t * VV, logh);
  }
}
